// EfficientTemporalAttention_28432683500086
// MI455X (gfx1250) — matrix operand order repaired and checked by reading, not compiled, not run on hardware
//
#include <hip/hip_runtime.h>
#include <hip/hip_bf16.h>

typedef __attribute__((ext_vector_type(16))) _Float16 v16h;
typedef __attribute__((ext_vector_type(8)))  _Float16 v8h;
typedef __attribute__((ext_vector_type(4)))  _Float16 v4h;
typedef __attribute__((ext_vector_type(8)))  float    v8f;

#define T_SEQ   2048
#define D_MODEL 512
#define NHEAD   8
#define HDIM    64
#define NBATCH  2

// ---------------- WMMA helper ----------------
__device__ __forceinline__ v8f wmma16(v16h a, v16h b, v8f c) {
    // D = A(16x32 f16) * B(32x16 f16) + C(16x16 f32)
    return __builtin_amdgcn_wmma_f32_16x16x32_f16(
        /*neg_a=*/false, a, /*neg_b=*/false, b,
        /*c_mod=*/(short)0, c, /*reuse_a=*/false, /*reuse_b=*/false);
}

// Load 16 contiguous f16 (32B) as v16h
__device__ __forceinline__ v16h load16(const _Float16* p) {
    v8h lo = *(const v8h*)p;
    v8h hi = *(const v8h*)(p + 8);
    v16h r;
#pragma unroll
    for (int i = 0; i < 8; i++) { r[i] = lo[i]; r[i + 8] = hi[i]; }
    return r;
}

// A-fragment (16x32, M x K) from row-major K-contiguous matrix, base at (m0,k0).
// Lane L: M = L%16. Lanes 0-15: K 0-15; lanes 16-31: K 16-31.
__device__ __forceinline__ v16h load_frag_a(const _Float16* base, int ld) {
    int lane = threadIdx.x & 31;
    const _Float16* p = base + (lane & 15) * ld + (lane >> 4) * 16;
    v8h lo = *(const v8h*)p;
    v8h hi = *(const v8h*)(p + 8);
    v16h r;
#pragma unroll
    for (int i = 0; i < 8; i++) { r[i] = lo[i]; r[i + 8] = hi[i]; }
    return r;
}

// B-fragment (32x16, K x N) for NT GEMM: B column n = row n of an N-major,
// K-contiguous matrix. Lane L: N = L%16; lanes 0-15 hold K 0-15, 16-31 hold K 16-31.
__device__ __forceinline__ v16h load_frag_b(const _Float16* base, int ld) {
    int lane = threadIdx.x & 31;
    return load16(base + (lane & 15) * ld + (lane >> 4) * 16);
}

// ---------------- fp32 -> fp16 conversion ----------------
__global__ void cvt4_kernel(const float4* __restrict__ s, v4h* __restrict__ d, int n4) {
    int i = blockIdx.x * blockDim.x + threadIdx.x;
    if (i < n4) {
        float4 v = s[i];
        v4h o;
        o[0] = (_Float16)v.x; o[1] = (_Float16)v.y;
        o[2] = (_Float16)v.z; o[3] = (_Float16)v.w;
        d[i] = o;
    }
}

// ------- fused QKV projection: 4 m-tiles x 3 matrices per wave (12 acc) -------
__global__ __launch_bounds__(128)
void qkv_gemm_kernel(const _Float16* __restrict__ x16,
                     const _Float16* __restrict__ wq16,
                     const _Float16* __restrict__ wk16,
                     const _Float16* __restrict__ wv16,
                     const float* __restrict__ bq,
                     const float* __restrict__ bk,
                     const float* __restrict__ bv,
                     _Float16* __restrict__ q16,   // (B,H,T,hd), pre-scaled
                     _Float16* __restrict__ k16,   // (B,H,T,hd)
                     _Float16* __restrict__ vt16)  // (B,H,hd,T) transposed
{
    const int wid  = threadIdx.x >> 5;
    const int lane = threadIdx.x & 31;
    const int m0 = (blockIdx.x * 4 + wid) * 64;   // 4 row tiles of 16 in (B*T)
    const int n0 = blockIdx.y * 16;               // col tile in D

    v8f accQ[4] = {{}, {}, {}, {}};
    v8f accK[4] = {{}, {}, {}, {}};
    v8f accV[4] = {{}, {}, {}, {}};

#pragma unroll 2
    for (int k0 = 0; k0 < D_MODEL; k0 += 32) {
        // B fragments loaded once, reused across 4 A fragments
        v16h bQ = load_frag_b(wq16 + n0 * D_MODEL + k0, D_MODEL);
        v16h bK = load_frag_b(wk16 + n0 * D_MODEL + k0, D_MODEL);
        v16h bV = load_frag_b(wv16 + n0 * D_MODEL + k0, D_MODEL);
#pragma unroll
        for (int tm = 0; tm < 4; tm++) {
            v16h a = load_frag_a(x16 + (m0 + tm * 16) * D_MODEL + k0, D_MODEL);
            accQ[tm] = wmma16(a, bQ, accQ[tm]);
            accK[tm] = wmma16(a, bK, accK[tm]);
            accV[tm] = wmma16(a, bV, accV[tm]);
        }
    }

    const int nloc = lane & 15, half = lane >> 4;
    const int n = n0 + nloc;
    const int h = n >> 6, d = n & 63;
    const float bqv = bq[n], bkv = bk[n], bvv = bv[n];

#pragma unroll
    for (int tm = 0; tm < 4; tm++) {
        const int mbase = m0 + tm * 16 + 8 * half;
        const int bb = mbase >> 11, tt = mbase & (T_SEQ - 1);
        const int qkbase = (((bb * NHEAD + h) * T_SEQ) + tt) * HDIM + d;
#pragma unroll
        for (int j = 0; j < 8; j++) {
            q16[qkbase + j * HDIM] = (_Float16)((accQ[tm][j] + bqv) * 0.125f); // hd^-0.5
            k16[qkbase + j * HDIM] = (_Float16)(accK[tm][j] + bkv);
        }
        v8h vv;
#pragma unroll
        for (int j = 0; j < 8; j++) vv[j] = (_Float16)(accV[tm][j] + bvv);
        *(v8h*)(vt16 + ((bb * NHEAD + h) * HDIM + d) * T_SEQ + tt) = vv; // t-contiguous
    }
}

// ---------------- windowed attention: 1 wave = 16 queries of one (b,h) --------
__global__ __launch_bounds__(128)
void attn_kernel(const _Float16* __restrict__ q16,
                 const _Float16* __restrict__ k16,
                 const _Float16* __restrict__ vt16,
                 _Float16* __restrict__ a16)     // (B,T,D) f16
{
    __shared__ _Float16 Pl[4][16 * 64];   // per-wave 16x64 P tile (cols 48-63 = 0 pad)

    const int wid  = threadIdx.x >> 5;
    const int lane = threadIdx.x & 31;
    const int tile = blockIdx.x * 4 + wid;          // 0..2047
    const int t0 = (tile & 127) * 16;
    const int bh = tile >> 7;                       // b*8+h
    const int b = bh >> 3, h = bh & 7;
    const int nloc = lane & 15, half = lane >> 4;

    // Q fragments (K = hd = 64 -> two 16x32 A-frags)
    const _Float16* qb = q16 + (bh * T_SEQ + t0) * HDIM;
    v16h aq0 = load_frag_a(qb,      HDIM);
    v16h aq1 = load_frag_a(qb + 32, HDIM);

    // S = Q K^T over 3 key tiles covering window [t0-16, t0+32)
    v8f s[3];
#pragma unroll
    for (int kt = 0; kt < 3; kt++) {
        int key = t0 - 16 + kt * 16 + nloc;
        int kc = key < 0 ? 0 : (key > T_SEQ - 1 ? T_SEQ - 1 : key);
        const _Float16* kr = k16 + (bh * T_SEQ + kc) * HDIM + half * 16;
        v16h b0 = load16(kr);       // hd 0..31 slice
        v16h b1 = load16(kr + 32);  // hd 32..63 slice
        v8f z = {};
        z = wmma16(aq0, b0, z);
        s[kt] = wmma16(aq1, b1, z);
    }

    // masked softmax (row = query, reduce across 16-lane half groups)
#pragma unroll
    for (int j = 0; j < 8; j++) {
        const int q = t0 + j + 8 * half;
#pragma unroll
        for (int kt = 0; kt < 3; kt++) {
            int key = t0 - 16 + kt * 16 + nloc;
            bool ok = (key >= q - 16) && (key < q + 16) && (key >= 0) && (key < T_SEQ);
            s[kt][j] = ok ? s[kt][j] : -1e30f;
        }
        float m = fmaxf(s[0][j], fmaxf(s[1][j], s[2][j]));
        m = fmaxf(m, __shfl_xor(m, 1, 32));
        m = fmaxf(m, __shfl_xor(m, 2, 32));
        m = fmaxf(m, __shfl_xor(m, 4, 32));
        m = fmaxf(m, __shfl_xor(m, 8, 32));
        float e0 = __expf(s[0][j] - m);
        float e1 = __expf(s[1][j] - m);
        float e2 = __expf(s[2][j] - m);
        float den = e0 + e1 + e2;
        den += __shfl_xor(den, 1, 32);
        den += __shfl_xor(den, 2, 32);
        den += __shfl_xor(den, 4, 32);
        den += __shfl_xor(den, 8, 32);
        float r = 1.0f / den;
        s[0][j] = e0 * r; s[1][j] = e1 * r; s[2][j] = e2 * r;
    }

    // P: C-layout -> A-fragment layout via per-wave LDS region (same-wave DS in order)
    _Float16* Pw = &Pl[wid][0];
#pragma unroll
    for (int j = 0; j < 8; j++) {
        const int row = j + 8 * half;
        Pw[row * 64 +      nloc] = (_Float16)s[0][j];
        Pw[row * 64 + 16 + nloc] = (_Float16)s[1][j];
        Pw[row * 64 + 32 + nloc] = (_Float16)s[2][j];
        Pw[row * 64 + 48 + nloc] = (_Float16)0.0f;   // zero pad keys 48..63
    }
    v16h pa0 = load_frag_a(Pw,      64);
    v16h pa1 = load_frag_a(Pw + 32, 64);

    // O = P V ; V is (B,H,hd,T) so B-frags are key-contiguous loads.
    // Edge blocks are fully masked (P==0), so whole-block address clamp is safe.
#pragma unroll
    for (int nt = 0; nt < 4; nt++) {
        const int d = nt * 16 + nloc;
        const _Float16* vr = vt16 + (bh * HDIM + d) * T_SEQ;
        int c0 = t0 - 16 + half * 16;  c0 = c0 < 0 ? 0 : c0;
        int c1 = t0 + 16 + half * 16;  c1 = c1 > T_SEQ - 16 ? T_SEQ - 16 : c1;
        v16h vb0 = load16(vr + c0);
        v16h vb1 = load16(vr + c1);
        v8f z = {};
        z = wmma16(pa0, vb0, z);
        v8f o = wmma16(pa1, vb1, z);
        const int col = h * HDIM + d;
#pragma unroll
        for (int j = 0; j < 8; j++)
            a16[(b * T_SEQ + t0 + j + 8 * half) * D_MODEL + col] = (_Float16)o[j];
    }
}

// ------- output projection: 4 m-tiles per wave sharing one B fragment ---------
__global__ __launch_bounds__(128)
void out_gemm_kernel(const _Float16* __restrict__ a16,
                     const _Float16* __restrict__ wo16,
                     const float* __restrict__ bo,
                     float* __restrict__ out)
{
    const int wid  = threadIdx.x >> 5;
    const int lane = threadIdx.x & 31;
    const int m0 = (blockIdx.x * 4 + wid) * 64;   // 4 row tiles of 16
    const int n0 = blockIdx.y * 16;

    v8f acc[4] = {{}, {}, {}, {}};
#pragma unroll 2
    for (int k0 = 0; k0 < D_MODEL; k0 += 32) {
        v16h b = load_frag_b(wo16 + n0 * D_MODEL + k0, D_MODEL);
#pragma unroll
        for (int tm = 0; tm < 4; tm++) {
            v16h a = load_frag_a(a16 + (m0 + tm * 16) * D_MODEL + k0, D_MODEL);
            acc[tm] = wmma16(a, b, acc[tm]);
        }
    }
    const int nloc = lane & 15, half = lane >> 4;
    const float bov = bo[n0 + nloc];
#pragma unroll
    for (int tm = 0; tm < 4; tm++)
#pragma unroll
        for (int j = 0; j < 8; j++)
            out[(m0 + tm * 16 + j + 8 * half) * D_MODEL + n0 + nloc] = acc[tm][j] + bov;
}

// ---------------- launch ----------------
extern "C" void kernel_launch(void* const* d_in, const int* in_sizes, int n_in,
                              void* d_out, int out_size, void* d_ws, size_t ws_size,
                              hipStream_t stream) {
    const float* x  = (const float*)d_in[0];
    const float* Wq = (const float*)d_in[1];
    const float* bq = (const float*)d_in[2];
    const float* Wk = (const float*)d_in[3];
    const float* bk = (const float*)d_in[4];
    const float* Wv = (const float*)d_in[5];
    const float* bv = (const float*)d_in[6];
    const float* Wo = (const float*)d_in[7];
    const float* bo = (const float*)d_in[8];
    float* out = (float*)d_out;

    char* ws = (char*)d_ws;
    const size_t MB = 1u << 20;
    _Float16* x16  = (_Float16*)(ws + 0);                    // 4 MiB
    _Float16* wq16 = (_Float16*)(ws + 4 * MB);               // 512 KiB
    _Float16* wk16 = (_Float16*)(ws + 4 * MB + 512 * 1024);
    _Float16* wv16 = (_Float16*)(ws + 5 * MB);
    _Float16* wo16 = (_Float16*)(ws + 5 * MB + 512 * 1024);
    _Float16* q16  = (_Float16*)(ws + 6 * MB);               // 4 MiB (B,H,T,hd)
    _Float16* k16  = (_Float16*)(ws + 10 * MB);              // 4 MiB (B,H,T,hd)
    _Float16* vt16 = (_Float16*)(ws + 14 * MB);              // 4 MiB (B,H,hd,T)
    _Float16* a16  = (_Float16*)(ws + 18 * MB);              // 4 MiB (B,T,D)

    // fp32 -> fp16 staging
    cvt4_kernel<<<2048, 256, 0, stream>>>((const float4*)x,  (v4h*)x16,  524288);
    cvt4_kernel<<<256,  256, 0, stream>>>((const float4*)Wq, (v4h*)wq16, 65536);
    cvt4_kernel<<<256,  256, 0, stream>>>((const float4*)Wk, (v4h*)wk16, 65536);
    cvt4_kernel<<<256,  256, 0, stream>>>((const float4*)Wv, (v4h*)wv16, 65536);
    cvt4_kernel<<<256,  256, 0, stream>>>((const float4*)Wo, (v4h*)wo16, 65536);

    // fused QKV projection: M=4096 in 64-row super-tiles, N=512 (32 n-tiles)
    qkv_gemm_kernel<<<dim3(16, 32), 128, 0, stream>>>(
        x16, wq16, wk16, wv16, bq, bk, bv, q16, k16, vt16);

    // windowed attention: 2048 query tiles, 4 waves per block
    attn_kernel<<<512, 128, 0, stream>>>(q16, k16, vt16, a16);

    // output projection
    out_gemm_kernel<<<dim3(16, 32), 128, 0, stream>>>(a16, wo16, bo, out);
}
